// SenseMemAct_40278203302327
// MI455X (gfx1250) — compile-verified, matrix operand order checked by reading, not run on hardware
//
#include <hip/hip_runtime.h>
#include <hip/hip_bf16.h>
#include <math.h>

#define B_SZ   64
#define L_SZ   1024
#define N_SZ   64
#define IN_DIM 128
#define MEM_SZ 1024
#define DEC_SZ 3

#define NWG    32              // persistent workgroups in recurrent kernel
#define JCOLS  (MEM_SZ / NWG)  // 32 h-columns per WG

typedef __bf16 bf16;
typedef __attribute__((ext_vector_type(16))) __bf16 v16bf;
typedef __attribute__((ext_vector_type(8)))  float  v8f;
typedef __attribute__((ext_vector_type(4)))  unsigned int u32x4;
typedef __attribute__((ext_vector_type(8)))  int i32x8;
typedef __attribute__((ext_vector_type(4)))  int i32x4;

// ---------------- workspace layout (bytes) ----------------
#define OFF_WFUSED 0u                                  // WfusedT [3072][64] bf16
#define OFF_BFUSED (OFF_WFUSED + 3072u*64u*2u)         // bfused [3072] f32
#define OFF_WMMT   (OFF_BFUSED + 3072u*4u)             // WmmT [2048][1024] bf16
#define OFF_XBF    (OFF_WMMT + 2048u*1024u*2u)         // x bf16 [64][1024][64]
#define OFF_H0     (OFF_XBF + 64u*1024u*64u*2u)        // h ping [64][1024] bf16
#define OFF_H1     (OFF_H0 + 64u*1024u*2u)             // h pong
#define OFF_CNT    (OFF_H1 + 64u*1024u*2u)             // barrier counters [1024] u32
#define OFF_PART   (OFF_CNT + 1024u*4u)                // partial logits [1024][NWG][64][3] f32

// ---------------- TDM: 1-D contiguous global -> LDS DMA ----------------
// Builds a Tensor DMA Descriptor (D#) per CDNA5 ISA §8.3/8.4:
//   group0: count=1 (valid), lds_addr[63:32], global_addr[120:64], type=2
//   group1: data_size=8B, tensor_dim0=tile_dim0=bytes/8, tensor_dim1=tile_dim1=1,
//           tensor_dim0_stride=bytes/8, everything else 0
__device__ __forceinline__ void tdm_copy_to_lds(unsigned lds_off, const void* gptr,
                                                unsigned bytes) {
    unsigned long long ga = (unsigned long long)gptr;
    unsigned n8 = bytes >> 3;                         // elements of 8 bytes
    u32x4 g0;
    g0.x = 1u;                                        // count=1, user descriptor
    g0.y = lds_off;                                   // lds_addr
    g0.z = (unsigned)(ga & 0xffffffffu);              // global_addr[31:0]
    g0.w = (unsigned)((ga >> 32) & 0x1ffffffu)        // global_addr[56:32]
         | (2u << 30);                                // type=2 ("image")
    i32x8 g1;
    g1[0] = (3 << 16);                                // data_size=8B; wg_mask=0
    g1[1] = (int)((n8 & 0xffffu) << 16);              // tensor_dim0[15:0] @bits63:48
    g1[2] = (int)((n8 >> 16) & 0xffffu) | (1 << 16);  // tensor_dim0[31:16]; tensor_dim1=1
    g1[3] = (int)(n8 << 16);                          // tile_dim0 @bits127:112
    g1[4] = 1;                                        // tile_dim1=1
    g1[5] = (int)n8;                                  // tensor_dim0_stride[31:0]
    g1[6] = 0;
    g1[7] = 0;
    i32x4 z4 = {0, 0, 0, 0};
#if defined(__clang_major__) && (__clang_major__ >= 23)
    i32x8 z8 = {0, 0, 0, 0, 0, 0, 0, 0};
    __builtin_amdgcn_tensor_load_to_lds(g0, g1, z4, z4, z8, 0);
#else
    __builtin_amdgcn_tensor_load_to_lds(g0, g1, z4, z4, 0);
#endif
}

// ---------------- prep kernels ----------------

// W_fused^T = (W_sense @ W_im)^T in bf16, plus b_fused = b_sense @ W_im (f32)
__global__ void k_prep_fuse(const float* __restrict__ Wsense, const float* __restrict__ bsense,
                            const float* __restrict__ Wim,
                            bf16* __restrict__ WfT, float* __restrict__ bfused) {
    int gid = blockIdx.x * blockDim.x + threadIdx.x;   // 3072*64 entries
    int c = gid >> 6, n = gid & 63;
    float acc = 0.f;
    for (int k = 0; k < IN_DIM; ++k)
        acc += Wsense[n * IN_DIM + k] * Wim[k * 3072 + c];
    WfT[c * 64 + n] = (bf16)acc;
    if (n == 0) {
        float bacc = 0.f;
        for (int k = 0; k < IN_DIM; ++k)
            bacc += bsense[k] * Wim[k * 3072 + c];
        bfused[c] = bacc;
    }
}

// W_mm^T in bf16: row c = column c of W_mm, contiguous over K
__global__ void k_prep_wmmt(const float* __restrict__ Wmm, bf16* __restrict__ WmmT) {
    int gid = blockIdx.x * blockDim.x + threadIdx.x;   // 2048*1024 entries
    int c = gid >> 10, k = gid & 1023;
    WmmT[c * MEM_SZ + k] = (bf16)Wmm[k * 2048 + c];
}

// x -> bf16
__global__ void k_prep_x(const float* __restrict__ x, bf16* __restrict__ xbf, int n) {
    int gid = blockIdx.x * blockDim.x + threadIdx.x;
    if (gid < n) xbf[gid] = (bf16)x[gid];
}

// zero h ping/pong + barrier counters (contiguous region)
__global__ void k_prep_zero(unsigned* __restrict__ p, int nwords) {
    int gid = blockIdx.x * blockDim.x + threadIdx.x;
    if (gid < nwords) p[gid] = 0u;
}

// ---------------- persistent recurrent kernel ----------------
__global__ __launch_bounds__(256, 1)
void k_recurrent(const bf16* __restrict__ WmmT, const bf16* __restrict__ WfT,
                 const float* __restrict__ bfused, const bf16* __restrict__ xbf,
                 const float* __restrict__ Wact,
                 bf16* __restrict__ h0, bf16* __restrict__ h1,
                 unsigned* __restrict__ cnt, float* __restrict__ partial) {
    __shared__ __attribute__((aligned(32))) bf16 sWmm[2 * JCOLS * MEM_SZ]; // 128 KB
    __shared__ __attribute__((aligned(32))) bf16 sWf[3 * JCOLS * N_SZ];    // 12 KB
    __shared__ float sRed[2][B_SZ][4];

    const int tid  = threadIdx.x;
    const int wg   = blockIdx.x;
    const int lane = tid & 31;
    const int wave = tid >> 5;
    const int mTile = wave >> 1;            // 0..3  (batch tile)
    const int nTile = wave & 1;             // 0..1  (column tile)
    const int half  = lane >> 4;            // K-half select: this code's contiguous k = 16*half + s order
    const int l15   = lane & 15;
    const int colLocal = nTile * 16 + l15;  // 0..31 within WG's column slice
    const int jcol  = wg * JCOLS + colLocal;
    const int arow  = mTile * 16 + l15;     // A-fragment row (batch index)
    const int kOff  = half * 16;

    // ---- stage this WG's weight slices into LDS via the Tensor Data Mover ----
    // Each region is contiguous in the pre-transposed workspace -> 1-D D# tiles.
    if (wave == 0) {
        unsigned ldsWmm = (unsigned)(unsigned long long)(uintptr_t)&sWmm[0];
        unsigned ldsWf  = (unsigned)(unsigned long long)(uintptr_t)&sWf[0];
        // m_a slice: WmmT rows [wg*32, wg*32+32)            -> sWmm[0      .. 64KB)
        tdm_copy_to_lds(ldsWmm,
                        WmmT + (size_t)(wg * JCOLS) * MEM_SZ, JCOLS * MEM_SZ * 2);
        // m_c slice: WmmT rows [1024+wg*32, 1024+wg*32+32)  -> sWmm[64KB .. 128KB)
        tdm_copy_to_lds(ldsWmm + JCOLS * MEM_SZ * 2,
                        WmmT + (size_t)(MEM_SZ + wg * JCOLS) * MEM_SZ, JCOLS * MEM_SZ * 2);
        // i_a / i_c / i_o slices of WfusedT (4 KB each)
        #pragma unroll
        for (int p = 0; p < 3; ++p)
            tdm_copy_to_lds(ldsWf + p * (JCOLS * N_SZ * 2),
                            WfT + (size_t)(p * MEM_SZ + wg * JCOLS) * N_SZ,
                            JCOLS * N_SZ * 2);
        __builtin_amdgcn_s_wait_tensorcnt(0);
    }
    __syncthreads();

    // per-lane static column data (column jcol is owned by this lane forever)
    const float bfa = bfused[jcol];
    const float bfc = bfused[MEM_SZ + jcol];
    const float bfo = bfused[2 * MEM_SZ + jcol];
    const float wa0 = Wact[jcol * 3 + 0];
    const float wa1 = Wact[jcol * 3 + 1];
    const float wa2 = Wact[jcol * 3 + 2];

    float hreg[8];   // persistent h state: element r == (b = mTile*16+half*8+r, j = jcol)
    #pragma unroll
    for (int r = 0; r < 8; ++r) hreg[r] = 0.f;

    const bf16* bA = sWmm + (size_t)colLocal * MEM_SZ + kOff;
    const bf16* bC = sWmm + (size_t)(JCOLS + colLocal) * MEM_SZ + kOff;

    for (int t = 0; t < L_SZ; ++t) {
        const bf16* hcur = (t & 1) ? h1 : h0;
        bf16*       hnxt = (t & 1) ? h0 : h1;

        // ---- recurrence GEMM: m_a, m_c (shared A fragment, two B streams) ----
        v8f acc_a = {}, acc_c = {};
        const bf16* hrow = hcur + (size_t)arow * MEM_SZ + kOff;
        #pragma unroll 4
        for (int kt = 0; kt < MEM_SZ / 32; ++kt) {
            v16bf af = *(const v16bf*)(hrow + kt * 32);
            v16bf ba = *(const v16bf*)(bA + kt * 32);
            v16bf bc = *(const v16bf*)(bC + kt * 32);
            acc_a = __builtin_amdgcn_wmma_f32_16x16x32_bf16(false, af, false, ba, (short)0, acc_a, false, false);
            acc_c = __builtin_amdgcn_wmma_f32_16x16x32_bf16(false, af, false, bc, (short)0, acc_c, false, false);
        }

        // ---- fused input projection: i_a, i_c, i_o = x_t @ W_fused (K = 64) ----
        v8f acc_ia = {}, acc_ic = {}, acc_io = {};
        const bf16* xrow = xbf + (size_t)arow * (L_SZ * N_SZ) + (size_t)t * N_SZ + kOff;
        #pragma unroll
        for (int kt = 0; kt < N_SZ / 32; ++kt) {
            v16bf ax = *(const v16bf*)(xrow + kt * 32);
            v16bf b0 = *(const v16bf*)(sWf + (size_t)(0 * JCOLS + colLocal) * N_SZ + kOff + kt * 32);
            v16bf b1 = *(const v16bf*)(sWf + (size_t)(1 * JCOLS + colLocal) * N_SZ + kOff + kt * 32);
            v16bf b2 = *(const v16bf*)(sWf + (size_t)(2 * JCOLS + colLocal) * N_SZ + kOff + kt * 32);
            acc_ia = __builtin_amdgcn_wmma_f32_16x16x32_bf16(false, ax, false, b0, (short)0, acc_ia, false, false);
            acc_ic = __builtin_amdgcn_wmma_f32_16x16x32_bf16(false, ax, false, b1, (short)0, acc_ic, false, false);
            acc_io = __builtin_amdgcn_wmma_f32_16x16x32_bf16(false, ax, false, b2, (short)0, acc_io, false, false);
        }

        if (t + 1 < L_SZ)   // pull next timestep's x slice toward the WGP
            __builtin_prefetch(xbf + (size_t)arow * (L_SZ * N_SZ) + (size_t)(t + 1) * N_SZ, 0, 1);

        // ---- gates, state update, on-the-fly actor partials ----
        #pragma unroll
        for (int r = 0; r < 8; ++r) {
            int brow = mTile * 16 + half * 8 + r;     // C-fragment row mapping
            float h  = hreg[r];
            float ag = 1.f + tanhf(acc_ia[r] + bfa + acc_a[r]);
            float cg = 1.f / (1.f + expf(-(acc_ic[r] + bfc + acc_c[r])));
            float hn = cg * h + (1.f - cg) * tanhf(acc_io[r] + bfo + ag * h);
            hreg[r] = hn;
            hnxt[(size_t)brow * MEM_SZ + jcol] = (bf16)hn;

            // reduce hn * W_act over the 16 columns held by this lane group
            float v0 = hn * wa0, v1 = hn * wa1, v2 = hn * wa2;
            #pragma unroll
            for (int m = 8; m >= 1; m >>= 1) {
                v0 += __shfl_xor(v0, m, 32);
                v1 += __shfl_xor(v1, m, 32);
                v2 += __shfl_xor(v2, m, 32);
            }
            if (l15 == 0) {
                sRed[nTile][brow][0] = v0;
                sRed[nTile][brow][1] = v1;
                sRed[nTile][brow][2] = v2;
            }
        }
        __syncthreads();
        if (tid < B_SZ * DEC_SZ) {
            int bb = tid / 3, d = tid % 3;
            partial[(((size_t)t * NWG + wg) * B_SZ + bb) * 3 + d] =
                sRed[0][bb][d] + sRed[1][bb][d];
        }

        // ---- device-wide step barrier (release h writes, acquire for t+1) ----
        __threadfence();
        __syncthreads();
        if (tid == 0) {
            atomicAdd(&cnt[t], 1u);
            while (atomicAdd(&cnt[t], 0u) < (unsigned)NWG)
                __builtin_amdgcn_s_sleep(1);
        }
        __syncthreads();
        __threadfence();
    }
}

// ---------------- actor reduce + softmax ----------------
__global__ void k_actor(const float* __restrict__ partial, const float* __restrict__ bact,
                        float* __restrict__ out) {
    int gid = blockIdx.x * blockDim.x + threadIdx.x;   // over B*L
    if (gid >= B_SZ * L_SZ) return;
    int b = gid / L_SZ, t = gid % L_SZ;
    float l0 = bact[0], l1 = bact[1], l2 = bact[2];
    const float* p = partial + (size_t)t * NWG * B_SZ * 3 + (size_t)b * 3;
    #pragma unroll 4
    for (int w = 0; w < NWG; ++w) {
        l0 += p[0]; l1 += p[1]; l2 += p[2];
        p += B_SZ * 3;
    }
    float mx = fmaxf(l0, fmaxf(l1, l2));
    float e0 = expf(l0 - mx), e1 = expf(l1 - mx), e2 = expf(l2 - mx);
    float inv = 1.f / (e0 + e1 + e2);
    float* o = out + ((size_t)b * L_SZ + t) * 3;
    o[0] = e0 * inv; o[1] = e1 * inv; o[2] = e2 * inv;
}

// ---------------- launch ----------------
extern "C" void kernel_launch(void* const* d_in, const int* in_sizes, int n_in,
                              void* d_out, int out_size, void* d_ws, size_t ws_size,
                              hipStream_t stream) {
    const float* x      = (const float*)d_in[0];
    const float* Wsense = (const float*)d_in[1];
    const float* bsense = (const float*)d_in[2];
    const float* Wim    = (const float*)d_in[3];
    const float* Wmm    = (const float*)d_in[4];
    const float* Wact   = (const float*)d_in[5];
    const float* bact   = (const float*)d_in[6];

    char* ws = (char*)d_ws;
    bf16*     WfT     = (bf16*)(ws + OFF_WFUSED);
    float*    bfused  = (float*)(ws + OFF_BFUSED);
    bf16*     WmmT    = (bf16*)(ws + OFF_WMMT);
    bf16*     xbf     = (bf16*)(ws + OFF_XBF);
    bf16*     h0      = (bf16*)(ws + OFF_H0);
    bf16*     h1      = (bf16*)(ws + OFF_H1);
    unsigned* cnt     = (unsigned*)(ws + OFF_CNT);
    float*    partial = (float*)(ws + OFF_PART);

    // weight fusion + bf16 conversion
    k_prep_fuse<<<(3072 * 64) / 256, 256, 0, stream>>>(Wsense, bsense, Wim, WfT, bfused);
    k_prep_wmmt<<<(2048 * 1024) / 256, 256, 0, stream>>>(Wmm, WmmT);
    int nx = B_SZ * L_SZ * N_SZ;
    k_prep_x<<<(nx + 255) / 256, 256, 0, stream>>>(x, xbf, nx);
    // zero h ping/pong + step counters (contiguous)
    int nz = (64 * 1024 * 2 * 2 + 1024 * 4) / 4;
    k_prep_zero<<<(nz + 255) / 256, 256, 0, stream>>>((unsigned*)(ws + OFF_H0), nz);

    // persistent recurrence (all NWG workgroups co-resident; ~145KB LDS each)
    k_recurrent<<<NWG, 256, 0, stream>>>(WmmT, WfT, bfused, xbf, Wact, h0, h1, cnt, partial);

    // actor reduce + softmax
    k_actor<<<(B_SZ * L_SZ + 255) / 256, 256, 0, stream>>>(partial, bact, (float*)d_out);
}
